// SphericalFourierNeuralOperatorModel_15908558864856
// MI455X (gfx1250) — matrix operand order repaired and checked by reading, not compiled, not run on hardware
//
#include <hip/hip_runtime.h>

// ---------------------------------------------------------------------------
// SFNO layer for MI455X (gfx1250): truncated-DFT-as-GEMM formulation.
// All heavy math through v_wmma_f32_16x16x32_bf16 (f32 accumulate).
// Memory-bound (~1.85 GB @ 23.3 TB/s); bf16 operands minimize traffic.
// Round-4: async builtin's params are int4-vector pointers (per round-3
// diagnostic) -> cast to v4i with explicit address spaces.
// ---------------------------------------------------------------------------

typedef __attribute__((ext_vector_type(16))) __bf16 v16bf;
typedef __attribute__((ext_vector_type(8)))  float  v8f;
typedef __attribute__((ext_vector_type(4)))  unsigned int u32x4;
typedef int vsi4 __attribute__((vector_size(16)));   // matches builtin param

typedef __attribute__((address_space(1))) vsi4* gv4p;
typedef __attribute__((address_space(3))) vsi4* lv4p;

union ABPack { u32x4 q[2]; v16bf v; };

#if __has_builtin(__builtin_amdgcn_global_load_async_to_lds_b128)
#define HAVE_ASYNC_LDS 1
#else
#define HAVE_ASYNC_LDS 0
#endif

// Opaque identity on a 32-bit index: blocks loop-invariant hoisting (and the
// 184-VGPR spills it caused) while keeping the base pointer's address-space
// provenance intact (ds_load for LDS, global_load for kernel args).
__device__ inline int opaque_i(int v) {
  asm volatile("" : "+v"(v));
  return v;
}

// A-matrix (16x32 bf16) lane layout: lanes 0-15 hold K={0..7,16..23},
// lanes 16-31 hold K={8..15,24..31}  (ISA 7.12.2)
__device__ inline v16bf load_a16(const __bf16* row, int kbase, int half) {
  ABPack p;
  p.q[0] = *(const u32x4*)(row + kbase + half * 8);
  p.q[1] = *(const u32x4*)(row + kbase + 16 + half * 8);
  return p.v;
}
// B-matrix (32x16 bf16) lane layout: lanes 0-15 K={0..7,16..23}, lanes 16-31 K={8..15,24..31}
__device__ inline v16bf load_b16(const __bf16* row, int kbase, int half) {
  ABPack p;
  const __bf16* s = row + kbase + half * 8;
  p.q[0] = *(const u32x4*)(s);
  p.q[1] = *(const u32x4*)(s + 16);
  return p.v;
}

__device__ inline v8f wmma_bf16(v16bf a, v16bf b, v8f c) {
  // (neg_a, A, neg_b, B, c_mod, C, reuse_a, reuse_b)
  return __builtin_amdgcn_wmma_f32_16x16x32_bf16(false, a, false, b, (short)0, c,
                                                 false, false);
}

#define V8F_ZERO {0.f,0.f,0.f,0.f,0.f,0.f,0.f,0.f}

// problem constants: B=4, CIN=COUT=128, H=360, W=720, 64x64 modes kept
#define KLON 736   // 720 padded to 23 k-steps of 32
#define KLAT 768   // 2 comps * 384 (360 padded)
#define AROW 744   // LDS row stride (bank-friendly: 372 words, gcd(52,64)=4)

// ---------------------------------------------------------------------------
// K1: build bf16 twiddle tables (forward/inverse DFT factors, ortho scaling,
// Hermitian doubling folded in).  Rebuilt every launch (deterministic, tiny).
// ---------------------------------------------------------------------------
__global__ void k_tables(__bf16* __restrict__ flonT, __bf16* __restrict__ flatA,
                         __bf16* __restrict__ ilatA, __bf16* __restrict__ filonT) {
  const int idx = blockIdx.x * 256 + threadIdx.x;
  const float TWO_PI = 6.28318530717958647692f;
  const float SFWD = rsqrtf(360.0f * 720.0f);
  const int N0 = 128 * KLON;        // forward lon, transposed [n][k]
  const int N1 = 128 * KLAT;        // forward lat A [(co,l)][(ct,hpad)]
  const int N2 = 720 * 128;         // inverse lat A [(cu*360+h)][(cy,l)]
  const int N3 = 720 * 128;         // inverse lon B^T [w][(cu,m)]

  if (idx < N0) {
    int n = idx / KLON, k = idx - n * KLON;
    float v = 0.f;
    if (k < 720) {
      int mm = n & 63;
      float th = TWO_PI * (float)((mm * k) % 720) / 720.f;
      v = (n < 64) ? __cosf(th) : -__sinf(th);   // re | im of e^{-i th}
    }
    flonT[n * KLON + k] = (__bf16)v;
  } else if (idx < N0 + N1) {
    int i = idx - N0;
    int row = i / KLAT, k = i - row * KLAT;
    int co = row >> 6, l = row & 63;
    int ct = k / 384, h = k - ct * 384;
    float v = 0.f;
    if (h < 360) {
      float th = TWO_PI * (float)((l * h) % 360) / 360.f;
      float cs = __cosf(th), sn = __sinf(th);
      // F = e^{-i th}: Fre = cs, Fim = -sn
      v = SFWD * ((co == 0) ? (ct == 0 ? cs : sn) : (ct == 0 ? -sn : cs));
    }
    flatA[row * KLAT + k] = (__bf16)v;
  } else if (idx < N0 + N1 + N2) {
    int i = idx - N0 - N1;
    int row = i >> 7, k = i & 127;            // row = cu*360+h
    int cu = row >= 360 ? 1 : 0, h = row - cu * 360;
    int cy = k >> 6, l = k & 63;
    float th = TWO_PI * (float)((l * h) % 360) / 360.f;
    float cs = __cosf(th), sn = __sinf(th);
    // U = sum Y * e^{+i th}
    float v = (cu == 0) ? (cy == 0 ? cs : -sn) : (cy == 0 ? sn : cs);
    ilatA[row * 128 + k] = (__bf16)v;
  } else if (idx < N0 + N1 + N2 + N3) {
    int i = idx - N0 - N1 - N2;
    int w = i >> 7, k = i & 127;
    int cu = k >> 6, mm = k & 63;
    float th = TWO_PI * (float)((mm * w) % 720) / 720.f;
    float sc = ((mm == 0) ? 1.f : 2.f) * SFWD; // Hermitian doubling + ortho
    float v = (cu == 0) ? sc * __cosf(th) : -sc * __sinf(th);
    filonT[w * 128 + k] = (__bf16)v;
  }
}

// ---------------------------------------------------------------------------
// K2: forward lon DFT.  Per (b,c) image: [360x720] x FlonT^T -> T1t (bf16).
// Whole FlonT B-panel (128x736 bf16 ~ 190KB) staged once into WGP LDS
// (async DMA via GLOBAL_LOAD_ASYNC_TO_LDS_B128 when available);
// x rows staged fp32->bf16 per 16-row M-tile.  Wave w owns N-tile w.
// T1t stored transposed [m][(ct*384+h)] so stage-3's B loads are contiguous.
// ---------------------------------------------------------------------------
__global__ void k_fwd_lon(const float* __restrict__ x,
                          const __bf16* __restrict__ flonT,
                          __bf16* __restrict__ t1t) {
  extern __shared__ __align__(16) __bf16 sm2[];
  __bf16* Ach  = sm2;               // 16  x AROW
  __bf16* Bpan = sm2 + 16 * AROW;   // 128 x AROW
  const int bc = blockIdx.x;
  const float* X = x + (size_t)bc * (360 * 720);
  __bf16* T = t1t + (size_t)bc * (64 * KLAT);
  const int tid = threadIdx.x, lane = tid & 31, wv = tid >> 5;
  const int half = lane >> 4, lr = lane & 15;
  const int n = wv * 16 + lr, ct = n >> 6, mcol = n & 63;

  // one-time: FlonT -> LDS in 16B chunks (rows: 92 chunks of 8 bf16)
  for (int i = tid; i < 128 * 92; i += 256) {
    int r = i / 92, c = i - r * 92;
    const __bf16* src = flonT + r * KLON + c * 8;
    __bf16* dst = Bpan + r * AROW + c * 8;
#if HAVE_ASYNC_LDS
    __builtin_amdgcn_global_load_async_to_lds_b128(
        (gv4p)(unsigned long long)src,
        (lv4p)(unsigned int)(unsigned long long)dst,
        0, 0);
#else
    *(u32x4*)dst = *(const u32x4*)src;
#endif
  }
#if HAVE_ASYNC_LDS
#if __has_builtin(__builtin_amdgcn_s_wait_asynccnt)
  __builtin_amdgcn_s_wait_asynccnt(0);
#else
  asm volatile("s_wait_asynccnt 0x0" ::: "memory");
#endif
#endif

  for (int mt = 0; mt < 23; ++mt) {
    __syncthreads();
    for (int i = tid; i < 16 * KLON; i += 256) {
      int r = i / KLON, k = i - r * KLON;
      int h = mt * 16 + r;
      float vx = (h < 360 && k < 720) ? X[h * 720 + k] : 0.f;
      Ach[r * AROW + k] = (__bf16)vx;
    }
    __syncthreads();
    v8f acc = V8F_ZERO;
    const int lro = opaque_i(lr);                 // re-derived every mt iter
    const __bf16* arow = &Ach[lro * AROW];
    const __bf16* brow = &Bpan[(wv * 16 + lro) * AROW];
#pragma unroll
    for (int ks = 0; ks < 23; ++ks)
      acc = wmma_bf16(load_a16(arow, ks * 32, half), load_b16(brow, ks * 32, half), acc);
#pragma unroll
    for (int r = 0; r < 8; ++r) {
      int h = mt * 16 + r + half * 8;
      if (h < 360) T[(size_t)mcol * KLAT + ct * 384 + h] = (__bf16)acc[r];
    }
  }
}

// ---------------------------------------------------------------------------
// K3: forward lat DFT (complex folded into real GEMM).
// D[(co,l)][m] = FlatA[128x768] * T1t[b,c]^T ; write Xs[l][m][co][b][c] (f32).
// FlatA is shared by all 512 blocks -> L2-resident; opaque_i() keeps the
// per-nt A-fragment loads in the loop instead of a 192-VGPR hoist+spill.
// ---------------------------------------------------------------------------
__global__ void k_fwd_lat(const __bf16* __restrict__ flatA,
                          const __bf16* __restrict__ t1t,
                          float* __restrict__ xs) {
  const int bc = blockIdx.x, b = bc >> 7, c = bc & 127;
  const __bf16* Tb = t1t + (size_t)bc * (64 * KLAT);
  const int tid = threadIdx.x, lane = tid & 31, wv = tid >> 5;
  const int half = lane >> 4, lr = lane & 15;
  for (int nt = 0; nt < 4; ++nt) {
    const int lro = opaque_i(lr);                 // re-derived every nt iter
    const __bf16* arow = flatA + (size_t)(wv * 16 + lro) * KLAT;
    const __bf16* brow = Tb + (size_t)(nt * 16 + lro) * KLAT;
    v8f acc = V8F_ZERO;
#pragma unroll
    for (int ks = 0; ks < 24; ++ks)
      acc = wmma_bf16(load_a16(arow, ks * 32, half), load_b16(brow, ks * 32, half), acc);
    int mcol = nt * 16 + lr;
#pragma unroll
    for (int r = 0; r < 8; ++r) {
      int row = wv * 16 + r + half * 8;   // (co,l)
      int co = row >> 6, l = row & 63;
      xs[((size_t)((l * 64 + mcol) * 2 + co) * 4 + b) * 128 + c] = acc[r];
    }
  }
}

// ---------------------------------------------------------------------------
// K4: per-mode complex channel mix.  One block per (l,m); consecutive blocks
// hit consecutive m -> weight cachelines (8 m per line) are reused out of the
// 192MB L2.  A = [[Xre,-Xim],[Xim,Xre]] (M=16 rows: (co,b), 8 used),
// B^T staged in LDS (bank-padded rows); K=256, N=128 (wave w -> o-tile w).
// ---------------------------------------------------------------------------
#define MIXROW 264   // 256 + 8 pad: 528B row stride, 16B aligned, bank-spread
__global__ void k_mix(const float* __restrict__ xs,
                      const float* __restrict__ wfft,
                      __bf16* __restrict__ yt) {
  extern __shared__ __align__(16) __bf16 sm[];
  __bf16* Alds = sm;                 // 16  x MIXROW
  __bf16* Blds = sm + 16 * MIXROW;   // 128 x MIXROW
  const int lm = blockIdx.x, l = lm >> 6, m = lm & 63;
  const int tid = threadIdx.x, lane = tid & 31, wv = tid >> 5;
  const int half = lane >> 4, lr = lane & 15;
  const float* Xm = xs + (size_t)lm * 1024;   // [co][b][c]

  for (int i = tid; i < 16 * 256; i += 256) {
    int r = i >> 8, k = i & 255;
    float v = 0.f;
    if (r < 8) {
      int co = r >> 2, bb = r & 3, cc = k >> 7, c = k & 127;
      float xre = Xm[bb * 128 + c];
      float xim = Xm[512 + bb * 128 + c];
      v = (co == 0) ? (cc == 0 ? xre : -xim) : (cc == 0 ? xim : xre);
    }
    Alds[r * MIXROW + k] = (__bf16)v;
  }
  for (int i = tid; i < 16384; i += 256) {
    int c = i >> 7, o = i & 127;
    const float* p = wfft + ((size_t)(c * 128 + o) * 4096 + lm) * 2;
    __builtin_prefetch(p + (size_t)2 * 128 * 4096 * 2, 0, 0);  // next c-block
    float2 wv2 = *(const float2*)p;
    Blds[o * MIXROW + c]       = (__bf16)wv2.x;   // Wre at k=c
    Blds[o * MIXROW + 128 + c] = (__bf16)wv2.y;   // Wim at k=128+c
  }
  __syncthreads();

  const __bf16* arow = Alds + lr * MIXROW;
  const __bf16* brow = Blds + (wv * 16 + lr) * MIXROW;
  v8f acc = V8F_ZERO;
#pragma unroll
  for (int ks = 0; ks < 8; ++ks)
    acc = wmma_bf16(load_a16(arow, ks * 32, half), load_b16(brow, ks * 32, half), acc);

  if (half == 0) {   // rows 0..7 hold (co,b); rows 8..15 are padding
    int o = wv * 16 + lr;
#pragma unroll
    for (int r = 0; r < 8; ++r) {
      int co = r >> 2, bb = r & 3;
      yt[((size_t)(bb * 128 + o) * 64 + m) * 128 + co * 64 + l] = (__bf16)acc[r];
    }
  }
}

// ---------------------------------------------------------------------------
// K5: inverse lat DFT.  Per (b,o): D[(cu*360+h)][m] = IlatA[720x128] * Yt^T.
// Write Ut[h][(cu*64+m)] bf16 (K-contiguous for K6's A loads).
// ---------------------------------------------------------------------------
__global__ void k_ilat(const __bf16* __restrict__ ilatA,
                       const __bf16* __restrict__ yt,
                       __bf16* __restrict__ ut) {
  const int bo = blockIdx.x;
  const __bf16* Y = yt + (size_t)bo * (64 * 128);
  __bf16* U = ut + (size_t)bo * (368 * 128);
  const int tid = threadIdx.x, lane = tid & 31, wv = tid >> 5;
  const int half = lane >> 4, lr = lane & 15;
  for (int t = wv; t < 180; t += 8) {           // 45 M-tiles x 4 N-tiles
    int mt = t >> 2, nt = t & 3;
    const __bf16* arow = ilatA + (size_t)(mt * 16 + lr) * 128;
    const __bf16* brow = Y + (size_t)(nt * 16 + lr) * 128;
    v8f acc = V8F_ZERO;
#pragma unroll
    for (int ks = 0; ks < 4; ++ks)
      acc = wmma_bf16(load_a16(arow, ks * 32, half), load_b16(brow, ks * 32, half), acc);
    int mcol = nt * 16 + lr;
#pragma unroll
    for (int r = 0; r < 8; ++r) {
      int row = mt * 16 + r + half * 8;         // cu*360 + h
      int cu = row >= 360 ? 1 : 0, h = row - cu * 360;
      U[(size_t)h * 128 + cu * 64 + mcol] = (__bf16)acc[r];
    }
  }
}

// ---------------------------------------------------------------------------
// K6: inverse lon DFT (Hermitian-doubled) -> final fp32 output.
// Per (b,o): out[360x720] = Ut[368x128] * FilonT^T ; 23x45 tiles per block.
// ---------------------------------------------------------------------------
__global__ void k_ilon(const __bf16* __restrict__ ut,
                       const __bf16* __restrict__ filonT,
                       float* __restrict__ out) {
  const int bo = blockIdx.x;
  const __bf16* U = ut + (size_t)bo * (368 * 128);
  float* O = out + (size_t)bo * (360 * 720);
  const int tid = threadIdx.x, lane = tid & 31, wv = tid >> 5;
  const int half = lane >> 4, lr = lane & 15;
  for (int t = wv; t < 23 * 45; t += 8) {
    int mt = t / 45, nt = t - mt * 45;
    const __bf16* arow = U + (size_t)(mt * 16 + lr) * 128;
    const __bf16* brow = filonT + (size_t)(nt * 16 + lr) * 128;
    v8f acc = V8F_ZERO;
#pragma unroll
    for (int ks = 0; ks < 4; ++ks)
      acc = wmma_bf16(load_a16(arow, ks * 32, half), load_b16(brow, ks * 32, half), acc);
    int wcol = nt * 16 + lr;
#pragma unroll
    for (int r = 0; r < 8; ++r) {
      int h = mt * 16 + r + half * 8;
      if (h < 360) O[(size_t)h * 720 + wcol] = acc[r];
    }
  }
}

// ---------------------------------------------------------------------------
extern "C" void kernel_launch(void* const* d_in, const int* in_sizes, int n_in,
                              void* d_out, int out_size, void* d_ws, size_t ws_size,
                              hipStream_t stream) {
  (void)in_sizes; (void)n_in; (void)out_size; (void)ws_size;
  const float* x    = (const float*)d_in[0];   // [4,128,360,720] f32
  const float* wfft = (const float*)d_in[1];   // [128,128,64,64,2] f32
  float* out = (float*)d_out;                  // [4,128,360,720] f32

  __bf16* ws = (__bf16*)d_ws;                  // workspace layout (bf16 elems)
  size_t o_flonT  = 0;
  size_t o_flatA  = o_flonT  + (size_t)128 * KLON;
  size_t o_ilatA  = o_flatA  + (size_t)128 * KLAT;
  size_t o_filonT = o_ilatA  + (size_t)720 * 128;
  size_t o_t1t    = o_filonT + (size_t)720 * 128;
  size_t o_yt     = o_t1t    + (size_t)512 * 64 * KLAT;
  size_t o_ut     = o_yt     + (size_t)512 * 64 * 128;
  size_t o_end    = o_ut     + (size_t)512 * 368 * 128;
  float* xs = (float*)(ws + ((o_end + 1) & ~(size_t)1));  // f32 [4096][2][4][128]

  dim3 blk(256);
  k_tables <<<1472, blk, 0, stream>>>(ws + o_flonT, ws + o_flatA,
                                      ws + o_ilatA, ws + o_filonT);
  size_t slon = (size_t)(16 * AROW + 128 * AROW) * sizeof(__bf16);  // ~214KB
  k_fwd_lon<<< 512, blk, slon, stream>>>(x, ws + o_flonT, ws + o_t1t);
  k_fwd_lat<<< 512, blk, 0, stream>>>(ws + o_flatA, ws + o_t1t, xs);
  size_t smix = (size_t)(16 * MIXROW + 128 * MIXROW) * sizeof(__bf16);
  k_mix    <<<4096, blk, smix, stream>>>(xs, wfft, ws + o_yt);
  k_ilat   <<< 512, blk, 0, stream>>>(ws + o_ilatA, ws + o_yt, ws + o_ut);
  k_ilon   <<< 512, blk, 0, stream>>>(ws + o_ut, ws + o_filonT, out);
}
